// WGAAE_75935021793689
// MI455X (gfx1250) — matrix operand order repaired and checked by reading, not compiled, not run on hardware
//
#include <hip/hip_runtime.h>
#include <cstdint>
#include <cstddef>

// ---------------------------------------------------------------------------
// Types for CDNA5 WMMA (wave32): v_wmma_f32_16x16x32_bf16
// ---------------------------------------------------------------------------
typedef __bf16 bf16_t;
typedef __attribute__((ext_vector_type(8)))  __bf16 v8bf;
typedef __attribute__((ext_vector_type(16))) __bf16 v16bf;
typedef __attribute__((ext_vector_type(8)))  float  v8f;

#define REAL_MIN 2.2e-10f

static constexpr int kN      = 50000;   // nodes  (50000 = 16*3125, tile-aligned)
static constexpr int kE      = 800000;  // edges
static constexpr int kET     = kE + kN; // edges + self loops
static constexpr int kHeads  = 8;
static constexpr int kDHead  = 16;      // 128/8

// ---------------------------------------------------------------------------
// GEMM: C[M,Nout] = A[M,K](bf16,row-major) * B(packed bf16) + bias
// One wave computes a 64x16 tile (4 stacked 16x16 WMMA tiles). Per k-step:
// load ONE B fragment + FOUR A fragments into *distinct* registers, then issue
// 4 back-to-back WMMAs sharing that B. Keeping all four A fragments live lets
// the scheduler overlap the tail loads with the first WMMAs (partial
// s_wait_loadcnt) instead of a full wait before every WMMA.
// A fragment (16-bit A 16x32 layout): lane<16 -> M=lane, elems0..7=K(k0..k0+7),
//   elems8..15=K(k0+16..k0+23); lane>=16 -> +8 on the K halves.
// Out-of-range M sub-tiles clamp their A row (redundant math, store skipped).
// ---------------------------------------------------------------------------
__global__ void __launch_bounds__(256)
gemm_bf16_wmma(const bf16_t* __restrict__ A, const bf16_t* __restrict__ Bp,
               const float* __restrict__ bias, float* __restrict__ C,
               int M, int K, int Nout)
{
    const int lane = threadIdx.x & 31;
    const int wave = threadIdx.x >> 5;
    const int tiles_m  = M >> 4;
    const int tiles_n  = Nout >> 4;
    const int tiles_m4 = (tiles_m + 3) >> 2;
    int t = blockIdx.x * 8 + wave;
    if (t >= tiles_m4 * tiles_n) return;
    const int tm0 = (t % tiles_m4) << 2;
    const int tn  = t / tiles_m4;
    const int half = lane >> 4;     // 0 or 1
    const int ln   = lane & 15;
    const int ktiles = K >> 5;

    const bf16_t* arow[4];
#pragma unroll
    for (int i = 0; i < 4; ++i) {
        int tm = tm0 + i;
        if (tm > tiles_m - 1) tm = tiles_m - 1;   // clamp (store suppressed later)
        arow[i] = A + (size_t)(tm * 16 + ln) * K + half * 8;
    }
    const bf16_t* bbase = Bp + ((size_t)tn * ktiles * 32 + lane) * 16;

    v8f acc[4] = { {}, {}, {}, {} };
    for (int k0 = 0; k0 < K; k0 += 32) {
        // Issue ALL loads first (distinct destinations) ...
        v16bf b = *(const v16bf*)(bbase + (size_t)(k0 >> 5) * 512);
        union { v8bf h8[2]; v16bf v; } ua[4];
#pragma unroll
        for (int i = 0; i < 4; ++i) {
            ua[i].h8[0] = *(const v8bf*)(arow[i] + k0);
            ua[i].h8[1] = *(const v8bf*)(arow[i] + k0 + 16);
        }
        // ... then the 4 WMMAs sharing B.
#pragma unroll
        for (int i = 0; i < 4; ++i)
            acc[i] = __builtin_amdgcn_wmma_f32_16x16x32_bf16(
                         false, ua[i].v, false, b, (short)0, acc[i], false, false);
    }
    const int col = tn * 16 + ln;
    const float bv = bias ? bias[col] : 0.0f;
#pragma unroll
    for (int i = 0; i < 4; ++i) {
        const int tm = tm0 + i;
        if (tm >= tiles_m) break;
        float* crow = C + (size_t)(tm * 16 + half * 8) * Nout + col;
#pragma unroll
        for (int r = 0; r < 8; ++r)
            crow[(size_t)r * Nout] = acc[i][r] + bv;
    }
}

// Pack f32 weight B[K,Nout] (row-major) into per-lane-contiguous bf16 fragments:
// element index = (((tn*ktiles)+kt)*32 + lane)*16 + e ; value = B[k, n]
// with n = tn*16 + lane%16 ; k = kt*32 + (lane/16)*8 + (e%8) + (e/8)*16
__global__ void pack_b_kernel(const float* __restrict__ B, bf16_t* __restrict__ Bp,
                              int K, int Nout)
{
    int idx = blockIdx.x * blockDim.x + threadIdx.x;
    int total = K * Nout;
    if (idx >= total) return;
    int e    = idx & 15;
    int lane = (idx >> 4) & 31;
    int rest = idx >> 9;
    int ktiles = K >> 5;
    int kt = rest % ktiles;
    int tn = rest / ktiles;
    int n = tn * 16 + (lane & 15);
    int k = kt * 32 + (lane >> 4) * 8 + (e & 7) + (e >> 3) * 16;
    Bp[idx] = (bf16_t)B[(size_t)k * Nout + n];
}

// ---------------------------------------------------------------------------
// Elementwise helpers
// ---------------------------------------------------------------------------
__global__ void f32_to_bf16_kernel(const float* __restrict__ in,
                                   bf16_t* __restrict__ out, size_t n)
{
    size_t i = (size_t)blockIdx.x * blockDim.x + threadIdx.x;
    if (i < n) out[i] = (bf16_t)in[i];
}

__global__ void fill_f32_kernel(float* __restrict__ p, float v, size_t n)
{
    size_t i = (size_t)blockIdx.x * blockDim.x + threadIdx.x;
    if (i < n) p[i] = v;
}

__global__ void combine_bias_kernel(const float* __restrict__ a,
                                    const float* __restrict__ b,
                                    float* __restrict__ out, int n)
{
    int i = blockIdx.x * blockDim.x + threadIdx.x;
    if (i < n) out[i] = a[i] + b[i];
}

// Per-node attention logits: al[n,h] = sum_d h[n, h*16+d] * a[h,d]
__global__ void attn_logits_kernel(const float* __restrict__ hp,
                                   const float* __restrict__ asrc,
                                   const float* __restrict__ adst,
                                   float* __restrict__ alsrc,
                                   float* __restrict__ aldst, int N)
{
    int i = blockIdx.x * blockDim.x + threadIdx.x;
    if (i >= N * kHeads) return;
    int h = i & 7, n = i >> 3;
    const float* hv = hp + (size_t)n * 128 + h * kDHead;
    float s = 0.f, d = 0.f;
#pragma unroll
    for (int j = 0; j < kDHead; ++j) {
        float v = hv[j];
        s += v * asrc[h * kDHead + j];
        d += v * adst[h * kDHead + j];
    }
    alsrc[i] = s;
    aldst[i] = d;
}

// ---------------------------------------------------------------------------
// Edge softmax (segment max / sum / weighted scatter-add)
// ---------------------------------------------------------------------------
__device__ __forceinline__ float leaky02(float v) { return v > 0.f ? v : 0.2f * v; }

__device__ __forceinline__ void atomic_max_f32(float* addr, float value)
{
    if (value >= 0.f)
        atomicMax((int*)addr, __float_as_int(value));
    else
        atomicMin((unsigned int*)addr, __float_as_uint(value));
}

__global__ void edge_max_kernel(const long long* __restrict__ ei, int E, int N,
                                const float* __restrict__ alsrc,
                                const float* __restrict__ aldst,
                                float* __restrict__ emax)
{
    int e = blockIdx.x * blockDim.x + threadIdx.x;
    if (e >= E + N) return;
    int s, d;
    if (e < E) { s = (int)ei[e]; d = (int)ei[E + e]; }
    else       { s = d = e - E; }
#pragma unroll
    for (int h = 0; h < kHeads; ++h) {
        float v = leaky02(alsrc[s * kHeads + h] + aldst[d * kHeads + h]);
        atomic_max_f32(&emax[d * kHeads + h], v);
    }
}

__global__ void edge_denom_kernel(const long long* __restrict__ ei, int E, int N,
                                  const float* __restrict__ alsrc,
                                  const float* __restrict__ aldst,
                                  const float* __restrict__ emax,
                                  float* __restrict__ denom)
{
    int e = blockIdx.x * blockDim.x + threadIdx.x;
    if (e >= E + N) return;
    int s, d;
    if (e < E) { s = (int)ei[e]; d = (int)ei[E + e]; }
    else       { s = d = e - E; }
#pragma unroll
    for (int h = 0; h < kHeads; ++h) {
        float v  = leaky02(alsrc[s * kHeads + h] + aldst[d * kHeads + h]);
        float ee = __expf(v - emax[d * kHeads + h]);
        atomicAdd(&denom[d * kHeads + h], ee);
    }
}

// One thread per (edge, head): z[dst, h*16..] += alpha * hproj[src, h*16..]
__global__ void edge_agg_kernel(const long long* __restrict__ ei, int E, int N,
                                const float* __restrict__ alsrc,
                                const float* __restrict__ aldst,
                                const float* __restrict__ emax,
                                const float* __restrict__ denom,
                                const float* __restrict__ hp,
                                float* __restrict__ z)
{
    int t = blockIdx.x * blockDim.x + threadIdx.x;
    if (t >= (E + N) * kHeads) return;
    int h = t & 7, e = t >> 3;
    int s, d;
    if (e < E) { s = (int)ei[e]; d = (int)ei[E + e]; }
    else       { s = d = e - E; }
    float v     = leaky02(alsrc[s * kHeads + h] + aldst[d * kHeads + h]);
    float ee    = __expf(v - emax[d * kHeads + h]);
    float alpha = ee / fmaxf(denom[d * kHeads + h], REAL_MIN);
    const float* hv = hp + (size_t)s * 128 + h * kDHead;
    float* zp       = z  + (size_t)d * 128 + h * kDHead;
#pragma unroll
    for (int j = 0; j < kDHead; ++j)
        atomicAdd(&zp[j], alpha * hv[j]);
}

// ---------------------------------------------------------------------------
// BatchNorm: per-column sum/sumsq via LDS atomics, then normalize (+bf16 copy)
// ---------------------------------------------------------------------------
__global__ void colstats_kernel(const float* __restrict__ z,
                                float* __restrict__ gsum, float* __restrict__ gsq,
                                int N)
{
    __shared__ float ss[128], sq[128];
    for (int i = threadIdx.x; i < 128; i += blockDim.x) { ss[i] = 0.f; sq[i] = 0.f; }
    __syncthreads();
    size_t total = (size_t)N * 128;
    for (size_t idx = (size_t)blockIdx.x * blockDim.x + threadIdx.x; idx < total;
         idx += (size_t)gridDim.x * blockDim.x) {
        float v = z[idx];
        int c = (int)(idx & 127);
        atomicAdd(&ss[c], v);
        atomicAdd(&sq[c], v * v);
    }
    __syncthreads();
    for (int i = threadIdx.x; i < 128; i += blockDim.x) {
        atomicAdd(&gsum[i], ss[i]);
        atomicAdd(&gsq[i],  sq[i]);
    }
}

__global__ void bn_norm_kernel(float* __restrict__ z, bf16_t* __restrict__ zbf,
                               const float* __restrict__ gsum,
                               const float* __restrict__ gsq,
                               const float* __restrict__ gamma,
                               const float* __restrict__ beta, int N)
{
    size_t idx = (size_t)blockIdx.x * blockDim.x + threadIdx.x;
    if (idx >= (size_t)N * 128) return;
    int c = (int)(idx & 127);
    float inv_n = 1.0f / (float)N;
    float mu  = gsum[c] * inv_n;
    float var = gsq[c] * inv_n - mu * mu;
    float v = (z[idx] - mu) * rsqrtf(var + 1e-5f) * gamma[c] + beta[c];
    z[idx]   = v;
    zbf[idx] = (bf16_t)v;
}

// ---------------------------------------------------------------------------
// Weibull sampling head: softplus/clip, lam/Gamma correction, 10-sample mean,
// transposed (K,N) output.
// ---------------------------------------------------------------------------
__device__ __forceinline__ unsigned int mix32(unsigned int a)
{
    a ^= a >> 16; a *= 0x7feb352dU;
    a ^= a >> 15; a *= 0x846ca68bU;
    a ^= a >> 16;
    return a;
}

__device__ __forceinline__ float softplusf(float x)
{
    return x > 20.0f ? x : log1pf(__expf(x));
}

__global__ void theta_kernel(const float* __restrict__ shp,
                             const float* __restrict__ scl,
                             float* __restrict__ out, int N, int K,
                             unsigned int layer)
{
    size_t idx = (size_t)blockIdx.x * blockDim.x + threadIdx.x;
    if (idx >= (size_t)N * K) return;
    int n = (int)(idx / K);
    int j = (int)(idx % K);
    float k   = softplusf(shp[idx]);
    k = fminf(fmaxf(k, 0.1f), 1000.0f);
    float lam = softplusf(scl[idx]);
    float invk = 1.0f / k;
    lam = fmaxf(lam, REAL_MIN) / __expf(lgammaf(1.0f + invk));
    float acc = 0.f;
#pragma unroll
    for (unsigned int s = 0; s < 10; ++s) {
        unsigned int h = mix32((unsigned int)n + 0x9E3779B9U *
                              ((unsigned int)j + 0x85EBCA6BU *
                               (s + 0xC2B2AE35U * layer + 1u)));
        float eps = ((float)(h >> 8) + 0.5f) * (1.0f / 16777216.0f);
        float t = -__logf(fmaxf(1.0f - eps, REAL_MIN));
        // t > 0, so pow(t, invk) = exp(invk * log(t))
        acc += lam * __expf(invk * __logf(t));
    }
    float th = acc * 0.1f;
    th = fminf(fmaxf(th, REAL_MIN), 1000.0f);
    out[(size_t)j * N + n] = th;
}

// ---------------------------------------------------------------------------
// Host-side sequencing
// ---------------------------------------------------------------------------
static inline int cdiv(long long a, long long b) { return (int)((a + b - 1) / b); }

static void run_gemm(const bf16_t* A, const float* W, const float* bias, float* C,
                     int M, int K, int Nout, bf16_t* bpack, hipStream_t s)
{
    int total = K * Nout;
    pack_b_kernel<<<cdiv(total, 256), 256, 0, s>>>(W, bpack, K, Nout);
    int tiles_m4 = ((M / 16) + 3) / 4;
    int tiles = tiles_m4 * (Nout / 16);
    gemm_bf16_wmma<<<cdiv(tiles, 8), 256, 0, s>>>(A, bpack, bias, C, M, K, Nout);
}

extern "C" void kernel_launch(void* const* d_in, const int* in_sizes, int n_in,
                              void* d_out, int out_size, void* d_ws, size_t ws_size,
                              hipStream_t stream)
{
    (void)in_sizes; (void)n_in; (void)out_size; (void)ws_size;
    const int N = kN, E = kE, ET = kET;

    // ---- inputs (setup_inputs dict order, lists flattened in order) ----
    const float*     x        = (const float*)d_in[0];
    const long long* ei       = (const long long*)d_in[1];
    // d_in[2], d_in[3]: phi (unused by the forward math)
    const float* gat_W[2]     = { (const float*)d_in[4],  (const float*)d_in[5]  };
    const float* a_src[2]     = { (const float*)d_in[6],  (const float*)d_in[7]  };
    const float* a_dst[2]     = { (const float*)d_in[8],  (const float*)d_in[9]  };
    const float* gat_b[2]     = { (const float*)d_in[10], (const float*)d_in[11] };
    const float* skip_W[2]    = { (const float*)d_in[12], (const float*)d_in[13] };
    const float* skip_b[2]    = { (const float*)d_in[14], (const float*)d_in[15] };
    const float* bn_gamma[2]  = { (const float*)d_in[16], (const float*)d_in[17] };
    const float* bn_beta[2]   = { (const float*)d_in[18], (const float*)d_in[19] };
    const float* fc_W[2]      = { (const float*)d_in[20], (const float*)d_in[21] };
    const float* fc_b[2]      = { (const float*)d_in[22], (const float*)d_in[23] };
    const float* shape_W[2]   = { (const float*)d_in[24], (const float*)d_in[25] };
    const float* shape_b[2]   = { (const float*)d_in[26], (const float*)d_in[27] };
    const float* scale_W[2]   = { (const float*)d_in[28], (const float*)d_in[29] };
    const float* scale_b[2]   = { (const float*)d_in[30], (const float*)d_in[31] };
    float* out = (float*)d_out;

    // ---- workspace carve-out (deterministic layout, 256B aligned) ----
    char* w = (char*)d_ws;
    auto carve = [&](size_t bytes) -> void* {
        void* p = (void*)w;
        w += (bytes + 255) & ~(size_t)255;
        return p;
    };
    bf16_t* xbf    = (bf16_t*)carve((size_t)N * 256 * 2);
    bf16_t* zbf0   = (bf16_t*)carve((size_t)N * 128 * 2);
    bf16_t* zbf1   = (bf16_t*)carve((size_t)N * 128 * 2);
    bf16_t* hidbf  = (bf16_t*)carve((size_t)N * 128 * 2);
    float*  hproj  = (float*)carve((size_t)N * 128 * 4);
    float*  z0     = (float*)carve((size_t)N * 128 * 4);
    float*  z1     = (float*)carve((size_t)N * 128 * 4);
    float*  hid    = (float*)carve((size_t)N * 128 * 4);
    float*  shp    = (float*)carve((size_t)N * 128 * 4);
    float*  scl    = (float*)carve((size_t)N * 128 * 4);
    float*  alsrc  = (float*)carve((size_t)N * kHeads * 4);
    float*  aldst  = (float*)carve((size_t)N * kHeads * 4);
    float*  emax   = (float*)carve((size_t)N * kHeads * 4);
    float*  denom  = (float*)carve((size_t)N * kHeads * 4);
    bf16_t* bpack  = (bf16_t*)carve((size_t)256 * 128 * 2);
    float*  cbias  = (float*)carve(128 * 4);
    float*  colsum = (float*)carve(128 * 4);
    float*  colsq  = (float*)carve(128 * 4);

    const size_t nh = (size_t)N * kHeads;
    const size_t nz = (size_t)N * 128;

    // Input features -> bf16 once.
    f32_to_bf16_kernel<<<cdiv((long long)N * 256, 256), 256, 0, stream>>>(x, xbf, (size_t)N * 256);

    const bf16_t* layer_in_bf[2] = { xbf, zbf0 };
    const int     layer_in_dim[2] = { 256, 128 };
    float*  zbuf[2]  = { z0, z1 };
    bf16_t* zbfb[2]  = { zbf0, zbf1 };

    // ================= GAT + skip + BN, two layers =================
    for (int l = 0; l < 2; ++l) {
        const bf16_t* Ain = layer_in_bf[l];
        const int Kdim = layer_in_dim[l];
        float* z = zbuf[l];

        // h = in @ gat_W  (no bias; gat_b folded into skip bias)
        run_gemm(Ain, gat_W[l], nullptr, hproj, N, Kdim, 128, bpack, stream);

        // attention logits per node/head
        attn_logits_kernel<<<cdiv((long long)nh, 256), 256, 0, stream>>>(
            hproj, a_src[l], a_dst[l], alsrc, aldst, N);

        // z initialized with skip path + (gat_b + skip_b)
        combine_bias_kernel<<<1, 128, 0, stream>>>(gat_b[l], skip_b[l], cbias, 128);
        run_gemm(Ain, skip_W[l], cbias, z, N, Kdim, 128, bpack, stream);

        // edge softmax: max -> denom -> weighted scatter-add into z
        fill_f32_kernel<<<cdiv((long long)nh, 256), 256, 0, stream>>>(emax, -INFINITY, nh);
        fill_f32_kernel<<<cdiv((long long)nh, 256), 256, 0, stream>>>(denom, 0.0f, nh);
        edge_max_kernel<<<cdiv(ET, 256), 256, 0, stream>>>(ei, E, N, alsrc, aldst, emax);
        edge_denom_kernel<<<cdiv(ET, 256), 256, 0, stream>>>(ei, E, N, alsrc, aldst, emax, denom);
        edge_agg_kernel<<<cdiv((long long)ET * kHeads, 256), 256, 0, stream>>>(
            ei, E, N, alsrc, aldst, emax, denom, hproj, z);

        // BatchNorm (batch stats), write normalized f32 + bf16 copy
        fill_f32_kernel<<<1, 128, 0, stream>>>(colsum, 0.0f, 128);
        fill_f32_kernel<<<1, 128, 0, stream>>>(colsq, 0.0f, 128);
        colstats_kernel<<<1024, 256, 0, stream>>>(z, colsum, colsq, N);
        bn_norm_kernel<<<cdiv((long long)nz, 256), 256, 0, stream>>>(
            z, zbfb[l], colsum, colsq, bn_gamma[l], bn_beta[l], N);
    }

    // ================= Weibull heads, two layers =================
    const int Kz[2] = { 128, 64 };
    size_t out_off = 0;
    for (int l = 0; l < 2; ++l) {
        const int Kl = Kz[l];
        // hid = hs[l] @ fc_W + fc_b     [N, Kl]
        run_gemm(zbfb[l], fc_W[l], fc_b[l], hid, N, 128, Kl, bpack, stream);
        f32_to_bf16_kernel<<<cdiv((long long)N * Kl, 256), 256, 0, stream>>>(
            hid, hidbf, (size_t)N * Kl);
        // shape / scale linears (raw; softplus fused into theta kernel)
        run_gemm(hidbf, shape_W[l], shape_b[l], shp, N, Kl, Kl, bpack, stream);
        run_gemm(hidbf, scale_W[l], scale_b[l], scl, N, Kl, Kl, bpack, stream);
        // theta (K,N) -> d_out
        theta_kernel<<<cdiv((long long)N * Kl, 256), 256, 0, stream>>>(
            shp, scl, out + out_off, N, Kl, (unsigned int)l);
        out_off += (size_t)Kl * N;
    }
}
